// BiLSTMCRFTagger_37400575213978
// MI455X (gfx1250) — matrix operand order repaired and checked by reading, not compiled, not run on hardware
//
#include <hip/hip_runtime.h>
#include <hip/hip_bf16.h>

// ---------------------------------------------------------------------------
// BiLSTM-CRF tagger for MI455X (gfx1250, wave32).
//  Phase 1: gather embeddings -> f16
//  Phase 2: input-gate GEMMs via v_wmma_f32_16x16x32_f16 (f32 accumulate)
//  Phase 3: recurrent scans (1 workgroup / direction, Whh^T streamed from L2)
//  Phase 4: tag projection + Viterbi decode
// ---------------------------------------------------------------------------

#define T_SEQ 2048
#define E_DIM 512
#define H_DIM 512
#define G_DIM 2048   // 4*H
#define NTAGS 12
#define START_TAG 10
#define STOP_TAG 11

typedef __attribute__((ext_vector_type(16))) _Float16 v16h;
typedef __attribute__((ext_vector_type(8)))  _Float16 v8h;
typedef __attribute__((ext_vector_type(8)))  float    v8f;

// ---------------------------------------------------------------------------
// Embedding gather + f32 -> f16 convert
// ---------------------------------------------------------------------------
__global__ void gather_embed_f16(const int* __restrict__ sent,
                                 const float* __restrict__ emb,
                                 _Float16* __restrict__ Xh) {
    int idx = blockIdx.x * blockDim.x + threadIdx.x;
    if (idx >= T_SEQ * E_DIM) return;
    int t = idx / E_DIM, e = idx - t * E_DIM;
    Xh[idx] = (_Float16)emb[(size_t)sent[t] * E_DIM + e];
}

__global__ void cvt_f32_to_f16(const float* __restrict__ src,
                               _Float16* __restrict__ dst, int n) {
    int idx = blockIdx.x * blockDim.x + threadIdx.x;
    if (idx < n) dst[idx] = (_Float16)src[idx];
}

// Whh [2][4H][H] -> WhhT [2][H][4H]  (coalesced recurrent matvec reads)
__global__ void transpose_whh(const float* __restrict__ Whh,
                              float* __restrict__ WT) {
    int idx = blockIdx.x * blockDim.x + threadIdx.x;
    if (idx >= 2 * G_DIM * H_DIM) return;
    int d   = idx / (G_DIM * H_DIM);
    int rem = idx - d * (G_DIM * H_DIM);
    int row = rem / H_DIM;            // gate row 0..4H-1
    int k   = rem - row * H_DIM;      // 0..H-1
    WT[(size_t)d * H_DIM * G_DIM + (size_t)k * G_DIM + row] = Whh[idx];
}

// ---------------------------------------------------------------------------
// WMMA GEMM: C[M][N] = A[M][K](f16) * W[N][K](f16)^T + b0[N] + b1[N]
// One wave -> 16x64 output strip (A-fragment reuse across 4 WMMAs).
// Fragments use contiguous k order (k = 16*(lane/16) + s), same for A and B (wave32).
// ---------------------------------------------------------------------------
__device__ __forceinline__ v16h load_a_frag(const _Float16* __restrict__ A,
                                            int lda, int m0, int k0, int lane) {
    const _Float16* p = A + (size_t)(m0 + (lane & 15)) * lda + k0 + ((lane < 16) ? 0 : 16);
    v8h lo = *(const v8h*)(p);
    v8h hi = *(const v8h*)(p + 8);
    v16h r;
#pragma unroll
    for (int i = 0; i < 8; ++i) { r[i] = lo[i]; r[8 + i] = hi[i]; }
    return r;
}

__device__ __forceinline__ v16h load_b_frag(const _Float16* __restrict__ W,
                                            int ldw, int n0, int k0, int lane) {
    const _Float16* p = W + (size_t)(n0 + (lane & 15)) * ldw + k0 + ((lane < 16) ? 0 : 16);
    v8h lo = *(const v8h*)(p);
    v8h hi = *(const v8h*)(p + 8);
    v16h r;
#pragma unroll
    for (int i = 0; i < 8; ++i) { r[i] = lo[i]; r[8 + i] = hi[i]; }
    return r;
}

__global__ void gemm_wmma_f16(const _Float16* __restrict__ A,
                              const _Float16* __restrict__ W,
                              const float* __restrict__ b0,
                              const float* __restrict__ b1,
                              float* __restrict__ C,
                              int M, int N, int K) {
    const int lane = threadIdx.x & 31;
    const int wave = blockIdx.x * (blockDim.x >> 5) + (threadIdx.x >> 5);
    const int wavesN = N >> 6;                 // 64 cols per wave
    const int wm = wave / wavesN;
    const int wn = wave - wm * wavesN;
    if (wm >= (M >> 4)) return;
    const int m0 = wm << 4, n0 = wn << 6;

    v8f acc0 = {}, acc1 = {}, acc2 = {}, acc3 = {};
    for (int k0 = 0; k0 < K; k0 += 32) {
        v16h a  = load_a_frag(A, K, m0, k0, lane);
        v16h bb0 = load_b_frag(W, K, n0 +  0, k0, lane);
        v16h bb1 = load_b_frag(W, K, n0 + 16, k0, lane);
        v16h bb2 = load_b_frag(W, K, n0 + 32, k0, lane);
        v16h bb3 = load_b_frag(W, K, n0 + 48, k0, lane);
        acc0 = __builtin_amdgcn_wmma_f32_16x16x32_f16(false, a, false, bb0, (short)0, acc0, false, false);
        acc1 = __builtin_amdgcn_wmma_f32_16x16x32_f16(false, a, false, bb1, (short)0, acc1, false, false);
        acc2 = __builtin_amdgcn_wmma_f32_16x16x32_f16(false, a, false, bb2, (short)0, acc2, false, false);
        acc3 = __builtin_amdgcn_wmma_f32_16x16x32_f16(false, a, false, bb3, (short)0, acc3, false, false);
    }
    // C/D layout: VGPR r <-> row m0 + r + (lane>=16 ? 8 : 0); col n0 + (lane&15)
    const int rbase = m0 + ((lane >> 4) << 3);
    const int cbase = n0 + (lane & 15);
    v8f* accs[4] = { &acc0, &acc1, &acc2, &acc3 };
#pragma unroll
    for (int i = 0; i < 4; ++i) {
        int col = cbase + (i << 4);
        float bias = b0[col] + b1[col];
#pragma unroll
        for (int r = 0; r < 8; ++r)
            C[(size_t)(rbase + r) * N + col] = (*accs[i])[r] + bias;
    }
}

// ---------------------------------------------------------------------------
// Recurrent scan: block 0 = forward, block 1 = backward. 512 threads.
// G: precomputed input gates + bias [T][4H]; WT: Whh^T [H][4H].
// Thread j owns hidden unit j (gate rows j, H+j, 2H+j, 3H+j).
// ---------------------------------------------------------------------------
__global__ void __launch_bounds__(512)
lstm_scan(const float* __restrict__ Gf, const float* __restrict__ Gb,
          const float* __restrict__ WTf, const float* __restrict__ WTb,
          float* __restrict__ Hout) {
    const int dir = blockIdx.x;
    const float* __restrict__ G  = dir ? Gb  : Gf;
    const float* __restrict__ WT = dir ? WTb : WTf;
    const int j = threadIdx.x;

    __shared__ float hsh[H_DIM];
    hsh[j] = 0.0f;
    float c = 0.0f;
    __syncthreads();

    for (int step = 0; step < T_SEQ; ++step) {
        const int t = dir ? (T_SEQ - 1 - step) : step;
        const float* __restrict__ g = G + (size_t)t * G_DIM;
        float ai = g[j], af = g[H_DIM + j], ag = g[2 * H_DIM + j], ao = g[3 * H_DIM + j];

        // prefetch next timestep's gate row while we grind the matvec
        if (step + 1 < T_SEQ) {
            const int tn = dir ? (t - 1) : (t + 1);
            __builtin_prefetch(G + (size_t)tn * G_DIM + j, 0, 0);
        }

#pragma unroll 4
        for (int k = 0; k < H_DIM; ++k) {
            const float hk = hsh[k];
            const float* __restrict__ w = WT + (size_t)k * G_DIM;
            ai = fmaf(w[j],             hk, ai);
            af = fmaf(w[H_DIM + j],     hk, af);
            ag = fmaf(w[2 * H_DIM + j], hk, ag);
            ao = fmaf(w[3 * H_DIM + j], hk, ao);
        }

        const float i_s = 1.0f / (1.0f + __expf(-ai));
        const float f_s = 1.0f / (1.0f + __expf(-af));
        const float o_s = 1.0f / (1.0f + __expf(-ao));
        c = f_s * c + i_s * tanhf(ag);
        const float h = o_s * tanhf(c);

        __syncthreads();               // everyone done reading old hsh
        hsh[j] = h;
        Hout[(size_t)t * (2 * H_DIM) + dir * H_DIM + j] = h;
        __syncthreads();               // new hsh visible
    }
}

// ---------------------------------------------------------------------------
// Tag projection: feats[t][tag] = H1[t][:] . Wtag[tag][:] + btag[tag]
// ---------------------------------------------------------------------------
__global__ void feats_kernel(const float* __restrict__ H1,
                             const float* __restrict__ Wtag,
                             const float* __restrict__ btag,
                             float* __restrict__ feats) {
    int idx = blockIdx.x * blockDim.x + threadIdx.x;
    if (idx >= T_SEQ * NTAGS) return;
    int t = idx / NTAGS, tag = idx - t * NTAGS;
    const float* __restrict__ h = H1 + (size_t)t * (2 * H_DIM);
    const float* __restrict__ w = Wtag + (size_t)tag * (2 * H_DIM);
    float s = btag[tag];
#pragma unroll 8
    for (int k = 0; k < 2 * H_DIM; ++k) s = fmaf(h[k], w[k], s);
    feats[idx] = s;
}

// ---------------------------------------------------------------------------
// Viterbi decode (12 tags) — single wave, sequential.
// trans[next*12+prev]; first-max tie-breaking matches jnp.argmax.
// ---------------------------------------------------------------------------
__global__ void viterbi_kernel(const float* __restrict__ feats,
                               const float* __restrict__ trans,
                               int* __restrict__ bptr,
                               float* __restrict__ out) {
    __shared__ float fv[NTAGS], nfv[NTAGS];
    const int n = threadIdx.x;
    if (n < NTAGS) fv[n] = (n == START_TAG) ? 0.0f : -10000.0f;
    __syncthreads();
    for (int t = 0; t < T_SEQ; ++t) {
        if (n < NTAGS) {
            float best = -3.0e38f; int arg = 0;
#pragma unroll
            for (int p = 0; p < NTAGS; ++p) {
                float s = fv[p] + trans[n * NTAGS + p];
                if (s > best) { best = s; arg = p; }
            }
            nfv[n] = best + feats[t * NTAGS + n];
            bptr[t * NTAGS + n] = arg;
        }
        __syncthreads();
        if (n < NTAGS) fv[n] = nfv[n];
        __syncthreads();
    }
    if (n == 0) {
        int bl = 0; float bs = -3.0e38f;
        for (int q = 0; q < NTAGS; ++q) {
            float s = fv[q] + trans[STOP_TAG * NTAGS + q];
            if (s > bs) { bs = s; bl = q; }
        }
        out[0] = bs;
        out[1 + (T_SEQ - 1)] = (float)bl;
        int cur = bl;
        for (int t = T_SEQ - 1; t >= 1; --t) {
            cur = bptr[t * NTAGS + cur];
            out[1 + (t - 1)] = (float)cur;
        }
    }
}

// ---------------------------------------------------------------------------
// Host launcher
// ---------------------------------------------------------------------------
extern "C" void kernel_launch(void* const* d_in, const int* in_sizes, int n_in,
                              void* d_out, int out_size, void* d_ws, size_t ws_size,
                              hipStream_t stream) {
    const int*   sent  = (const int*)  d_in[0];
    const float* emb   = (const float*)d_in[1];
    const float* Wih0  = (const float*)d_in[2];   // [2][2048][512]
    const float* Whh0  = (const float*)d_in[3];   // [2][2048][512]
    const float* bih0  = (const float*)d_in[4];   // [2][2048]
    const float* bhh0  = (const float*)d_in[5];
    const float* Wih1  = (const float*)d_in[6];   // [2][2048][1024]
    const float* Whh1  = (const float*)d_in[7];   // [2][2048][512]
    const float* bih1  = (const float*)d_in[8];
    const float* bhh1  = (const float*)d_in[9];
    const float* Wtag  = (const float*)d_in[10];  // [12][1024]
    const float* btag  = (const float*)d_in[11];
    const float* trans = (const float*)d_in[12];  // [12][12]
    float* out = (float*)d_out;

    // ---- carve workspace -------------------------------------------------
    char* base = (char*)d_ws;
    size_t off = 0;
    auto alloc = [&](size_t bytes) -> char* {
        char* p = base + off;
        off = (off + bytes + 255) & ~(size_t)255;
        return p;
    };
    _Float16* Xh    = (_Float16*)alloc((size_t)T_SEQ * E_DIM * 2);          // 2 MB
    _Float16* Wih0h = (_Float16*)alloc((size_t)2 * G_DIM * E_DIM * 2);      // 4 MB
    _Float16* Wih1h = (_Float16*)alloc((size_t)2 * G_DIM * 2 * H_DIM * 2);  // 8 MB
    float*    WT0   = (float*)   alloc((size_t)2 * H_DIM * G_DIM * 4);      // 8 MB
    float*    WT1   = (float*)   alloc((size_t)2 * H_DIM * G_DIM * 4);      // 8 MB
    float*    Gbuf  = (float*)   alloc((size_t)2 * T_SEQ * G_DIM * 4);      // 32 MB (reused L0/L1)
    float*    H0    = (float*)   alloc((size_t)T_SEQ * 2 * H_DIM * 4);      // 8 MB
    _Float16* H0h   = (_Float16*)alloc((size_t)T_SEQ * 2 * H_DIM * 2);      // 4 MB
    float*    H1    = (float*)   alloc((size_t)T_SEQ * 2 * H_DIM * 4);      // 8 MB
    float*    feats = (float*)   alloc((size_t)T_SEQ * NTAGS * 4);
    int*      bptr  = (int*)     alloc((size_t)T_SEQ * NTAGS * 4);

    float* Gf = Gbuf;
    float* Gb = Gbuf + (size_t)T_SEQ * G_DIM;

    // ---- phase 1: gather + conversions + transpose -----------------------
    gather_embed_f16<<<(T_SEQ * E_DIM + 255) / 256, 256, 0, stream>>>(sent, emb, Xh);
    cvt_f32_to_f16<<<(2 * G_DIM * E_DIM + 255) / 256, 256, 0, stream>>>(Wih0, Wih0h, 2 * G_DIM * E_DIM);
    cvt_f32_to_f16<<<(2 * G_DIM * 2 * H_DIM + 255) / 256, 256, 0, stream>>>(Wih1, Wih1h, 2 * G_DIM * 2 * H_DIM);
    transpose_whh<<<(2 * G_DIM * H_DIM + 255) / 256, 256, 0, stream>>>(Whh0, WT0);
    transpose_whh<<<(2 * G_DIM * H_DIM + 255) / 256, 256, 0, stream>>>(Whh1, WT1);

    // ---- phase 2: layer-0 input GEMMs (WMMA) -----------------------------
    // waves = (M/16)*(N/64) = 128*32 = 4096; 8 waves/block -> 512 blocks
    const int gemmBlocks = (T_SEQ / 16) * (G_DIM / 64) / 8;
    gemm_wmma_f16<<<gemmBlocks, 256, 0, stream>>>(Xh, Wih0h,
                                                  bih0, bhh0, Gf,
                                                  T_SEQ, G_DIM, E_DIM);
    gemm_wmma_f16<<<gemmBlocks, 256, 0, stream>>>(Xh, Wih0h + (size_t)G_DIM * E_DIM,
                                                  bih0 + G_DIM, bhh0 + G_DIM, Gb,
                                                  T_SEQ, G_DIM, E_DIM);

    // ---- phase 3a: layer-0 recurrent scan --------------------------------
    lstm_scan<<<2, H_DIM, 0, stream>>>(Gf, Gb, WT0, WT0 + (size_t)H_DIM * G_DIM, H0);

    // ---- phase 2': layer-1 input GEMMs -----------------------------------
    cvt_f32_to_f16<<<(T_SEQ * 2 * H_DIM + 255) / 256, 256, 0, stream>>>(H0, H0h, T_SEQ * 2 * H_DIM);
    gemm_wmma_f16<<<gemmBlocks, 256, 0, stream>>>(H0h, Wih1h,
                                                  bih1, bhh1, Gf,
                                                  T_SEQ, G_DIM, 2 * H_DIM);
    gemm_wmma_f16<<<gemmBlocks, 256, 0, stream>>>(H0h, Wih1h + (size_t)G_DIM * 2 * H_DIM,
                                                  bih1 + G_DIM, bhh1 + G_DIM, Gb,
                                                  T_SEQ, G_DIM, 2 * H_DIM);

    // ---- phase 3b: layer-1 recurrent scan --------------------------------
    lstm_scan<<<2, H_DIM, 0, stream>>>(Gf, Gb, WT1, WT1 + (size_t)H_DIM * G_DIM, H1);

    // ---- phase 4: tag projection + Viterbi -------------------------------
    feats_kernel<<<(T_SEQ * NTAGS + 255) / 256, 256, 0, stream>>>(H1, Wtag, btag, feats);
    viterbi_kernel<<<1, 32, 0, stream>>>(feats, trans, bptr, out);
}
